// STDefACMixAttn_79207786873029
// MI455X (gfx1250) — matrix operand order repaired and checked by reading, not compiled, not run on hardware
//
#include <hip/hip_runtime.h>
#include <math.h>
#include <stdint.h>

typedef __attribute__((ext_vector_type(16))) _Float16 v16h;
typedef __attribute__((ext_vector_type(8)))  float    v8f;

#if __has_builtin(__builtin_amdgcn_global_load_async_to_lds_b128) && \
    __has_builtin(__builtin_amdgcn_s_wait_asynccnt)
#define STDA_ASYNC_LDS 1
#endif

// b128 async transfers are typed as 4x i32 vectors (per the clang signature:
// global AS1 v4i* src, shared AS3 v4i* dst, imm offset, imm cpol).
typedef int v4i_ __attribute__((vector_size(16)));
typedef __attribute__((address_space(1))) v4i_ as1_v4i;
typedef __attribute__((address_space(3))) v4i_ as3_v4i;

namespace {
constexpr int B_    = 4;
constexpr int NIMG_ = 4;
constexpr int H_    = 80;
constexpr int W_    = 80;
constexpr int C_    = 256;
constexpr int HW_   = H_ * W_;       // 6400
constexpr int NTOK_ = NIMG_ * HW_;   // 25600
constexpr int LR_   = 200;           // real 3D keys (2*10*10)
constexpr int LP_   = 224;           // padded to 7 chunks of K=32
}

// ---------------- WMMA fragment index helpers (wave32, 16x16x32 f16) -------
// A (16x32, 16-bit): lanes 0-15 -> M=lane, K in {0-7,16-23}; lanes 16-31 ->
// M=lane-16, K in {8-15,24-31}.
__device__ __forceinline__ int amapA(int lane, int j) {
  return ((lane >> 4) << 3) + ((j >> 3) << 4) + (j & 7);
}
// B (32x16, 16-bit): N = lane%16; lanes 0-15 K in {0-7,16-23}; lanes 16-31 K in {8-15,24-31}.
__device__ __forceinline__ int bmapK(int lane, int j) {
  return ((lane >> 4) << 3) + ((j >> 3) << 4) + (j & 7);
}
__device__ __forceinline__ v8f wmma16(v16h a, v16h b, v8f c) {
  return __builtin_amdgcn_wmma_f32_16x16x32_f16(false, a, false, b, (short)0, c,
                                                false, false);
}
__device__ __forceinline__ float geluf(float x) {
  return 0.5f * x * (1.0f + tanhf(0.7978845608028654f * (x + 0.044715f * x * x * x)));
}

// ---------------- weight f16 conversion -------------------------------------
__global__ void k_wconv(const float* Wqkv, const float* Wout3,
                        _Float16* wqkvh, _Float16* wout3h) {
  int i = blockIdx.x * blockDim.x + threadIdx.x;
  if (i < 256 * 64) wqkvh[i] = (_Float16)Wqkv[i];
  if (i < 256 * 32) wout3h[i] = (_Float16)Wout3[i];
}

// ---------------- 1) QKV GEMM (WMMA): [102400 x 256] * [256 x 64] -----------
__global__ __launch_bounds__(32) void k_qkv(const float* __restrict__ x,
                                            const _Float16* __restrict__ wqkvh,
                                            const float* __restrict__ bqkv,
                                            float* __restrict__ qkv) {
  const int lane = threadIdx.x;
  const int row0 = blockIdx.x * 16;
  const int m16 = lane & 15;
  v8f acc[4];
  for (int t = 0; t < 4; ++t)
    for (int e = 0; e < 8; ++e) acc[t][e] = 0.0f;
  const int r = row0 + m16;
  const int bidx = r / NTOK_;
  const int tt = r - bidx * NTOK_;
  const float* xrow = x + ((size_t)tt * B_ + bidx) * C_;  // x[t][b][:]
  for (int kc = 0; kc < 8; ++kc) {
    if (kc < 7) __builtin_prefetch(xrow + (kc + 1) * 32, 0, 1);
    v16h a;
    for (int j = 0; j < 16; ++j) a[j] = (_Float16)xrow[kc * 32 + amapA(lane, j)];
    for (int nt = 0; nt < 4; ++nt) {
      v16h bf;
      const int col = nt * 16 + m16;
      for (int j = 0; j < 16; ++j)
        bf[j] = wqkvh[(kc * 32 + bmapK(lane, j)) * 64 + col];
      acc[nt] = wmma16(a, bf, acc[nt]);
    }
  }
  const int mb = (lane >> 4) * 8;
  for (int nt = 0; nt < 4; ++nt) {
    const int col = nt * 16 + m16;
    const float bb = bqkv[col];
    for (int rI = 0; rI < 8; ++rI)
      qkv[(size_t)(row0 + mb + rI) * 64 + col] = acc[nt][rI] + bb;
  }
}

// ---------------- 2) fin = W_fc mix of qkv -> [16,36,H,W] -------------------
__global__ void k_fin(const float* __restrict__ qkv, const float* __restrict__ Wfc,
                      const float* __restrict__ bfc, float* __restrict__ fin) {
  int gid = blockIdx.x * blockDim.x + threadIdx.x;
  if (gid >= B_ * NIMG_ * HW_) return;
  int bn = gid / HW_;
  int hw = gid - bn * HW_;
  int b = bn / NIMG_, n = bn - b * NIMG_;
  const float* q = qkv + ((size_t)b * NTOK_ + n * HW_ + hw) * 64;
  float ql[64];
  for (int i = 0; i < 64; ++i) ql[i] = q[i];
  for (int o = 0; o < 9; ++o)
    for (int k = 0; k < 4; ++k) {
      float s = bfc[o];
      for (int c = 0; c < 16; ++c) s += Wfc[o * 16 + c] * ql[c * 4 + k];
      fin[((size_t)bn * 36 + (k * 9 + o)) * HW_ + hw] = s;
    }
}

// ---------------- 3) 3x3 offset conv: fin(36) -> off(18) --------------------
__global__ void k_offconv(const float* __restrict__ fin, const float* __restrict__ Wd,
                          const float* __restrict__ bd, float* __restrict__ offb) {
  int gid = blockIdx.x * blockDim.x + threadIdx.x;
  if (gid >= 16 * 18 * HW_) return;
  int bn = gid / (18 * HW_);
  int rem = gid - bn * 18 * HW_;
  int oc = rem / HW_;
  int hw = rem - oc * HW_;
  int h = hw / W_, w = hw - h * W_;
  float acc = bd[oc];
  for (int ic = 0; ic < 36; ++ic) {
    const float* ch = fin + ((size_t)bn * 36 + ic) * HW_;
    const float* wk = Wd + ((size_t)oc * 36 + ic) * 9;
    for (int ky = 0; ky < 3; ++ky) {
      int y = h + ky - 1;
      if (y < 0 || y >= H_) continue;
      for (int kx = 0; kx < 3; ++kx) {
        int xx = w + kx - 1;
        if (xx < 0 || xx >= W_) continue;
        acc += ch[y * W_ + xx] * wk[ky * 3 + kx];
      }
    }
  }
  offb[((size_t)bn * 18 + oc) * HW_ + hw] = acc;
}

// ---------------- 4) deformable sample + depthwise + W_st -> rate2*conv -----
__global__ void k_depst(const float* __restrict__ fin, const float* __restrict__ offb,
                        const float* __restrict__ Wdep, const float* __restrict__ bdep,
                        const float* __restrict__ Wst, const float* __restrict__ bst,
                        const float* __restrict__ rate2p, float* __restrict__ out) {
  int gid = blockIdx.x * blockDim.x + threadIdx.x;
  if (gid >= B_ * NIMG_ * HW_ * 4) return;
  int g = gid & 3;
  int pix = gid >> 2;
  int bn = pix / HW_;
  int hw = pix - bn * HW_;
  int h = hw / W_, w = hw - h * W_;
  int b = bn / NIMG_, n = bn - b * NIMG_;
  float dep[27];
  for (int o2 = 0; o2 < 27; ++o2) dep[o2] = bdep[g * 27 + o2];
  for (int tap = 0; tap < 9; ++tap) {
    float oy = offb[((size_t)bn * 18 + tap * 2 + 0) * HW_ + hw];
    float ox = offb[((size_t)bn * 18 + tap * 2 + 1) * HW_ + hw];
    float py = (float)h + (float)(tap / 3 - 1) + oy;
    float px = (float)w + (float)(tap % 3 - 1) + ox;
    float y0f = floorf(py), x0f = floorf(px);
    int iy0 = (int)y0f, ix0 = (int)x0f;
    float wy = py - y0f, wx = px - x0f;
    float w00 = (1 - wy) * (1 - wx), w01 = (1 - wy) * wx;
    float w10 = wy * (1 - wx), w11 = wy * wx;
    bool vy0 = (iy0 >= 0 && iy0 < H_), vy1 = (iy0 + 1 >= 0 && iy0 + 1 < H_);
    bool vx0 = (ix0 >= 0 && ix0 < W_), vx1 = (ix0 + 1 >= 0 && ix0 + 1 < W_);
    int cy0 = max(0, min(H_ - 1, iy0)), cy1 = max(0, min(H_ - 1, iy0 + 1));
    int cx0 = max(0, min(W_ - 1, ix0)), cx1 = max(0, min(W_ - 1, ix0 + 1));
    for (int i = 0; i < 9; ++i) {
      const float* ch = fin + ((size_t)bn * 36 + g * 9 + i) * HW_;
      float s = 0.0f;
      if (vy0 && vx0) s += ch[cy0 * W_ + cx0] * w00;
      if (vy0 && vx1) s += ch[cy0 * W_ + cx1] * w01;
      if (vy1 && vx0) s += ch[cy1 * W_ + cx0] * w10;
      if (vy1 && vx1) s += ch[cy1 * W_ + cx1] * w11;
      const float* wgp = Wdep + (size_t)(g * 27) * 81 + i * 9 + tap;
      for (int o2 = 0; o2 < 27; ++o2) dep[o2] += s * wgp[o2 * 81];
    }
  }
  const float r2 = rate2p[0];
  const int t = n * HW_ + hw;
  float* orow = out + (size_t)t * B_ * C_ + b * C_ + g * 64;
  for (int o = 0; o < 64; ++o) {
    float acc = bst[g * 64 + o];
    const float* wr = Wst + (size_t)(g * 64 + o) * 27;
    for (int i = 0; i < 27; ++i) acc += dep[i] * wr[i];
    orow[o] = r2 * acc;
  }
}

// ---------------- 5a) value image for 2D attention --------------------------
__global__ void k_vimg(const float* __restrict__ qkv, const float* __restrict__ Wval,
                       const float* __restrict__ bval, float* __restrict__ vimg) {
  int gid = blockIdx.x * blockDim.x + threadIdx.x;
  if (gid >= B_ * NIMG_ * HW_) return;
  int bn = gid / HW_;
  int hw = gid - bn * HW_;
  int b = bn / NIMG_, n = bn - b * NIMG_;
  const float* k2 = qkv + ((size_t)b * NTOK_ + n * HW_ + hw) * 64 + 32;
  float kl[32];
  for (int c = 0; c < 32; ++c) kl[c] = k2[c];
  for (int d32 = 0; d32 < 32; ++d32) {
    float v = bval[d32];
    for (int c = 0; c < 32; ++c) v += kl[c] * Wval[c * 32 + d32];
    vimg[(size_t)(bn * 32 + d32) * HW_ + hw] = v;  // (bn,head,d) channel = d32
  }
}

// ---------------- 5b) 2D deformable attention + out proj --------------------
__global__ void k_attn2d(const float* __restrict__ qkv, const float* __restrict__ vimg,
                         const float* __restrict__ Woff, const float* __restrict__ boff,
                         const float* __restrict__ Watt, const float* __restrict__ batt,
                         const float* __restrict__ Wout2, const float* __restrict__ bout2,
                         float* __restrict__ a2out) {
  int gid = blockIdx.x * blockDim.x + threadIdx.x;
  if (gid >= B_ * NIMG_ * HW_) return;
  int bn = gid / HW_;
  int hw = gid - bn * HW_;
  int h = hw / W_, w = hw - h * W_;
  int b = bn / NIMG_, n = bn - b * NIMG_;
  const float* q = qkv + ((size_t)b * NTOK_ + n * HW_ + hw) * 64;
  float ql[32];
  for (int c = 0; c < 32; ++c) ql[c] = q[c];
  float a2[32];
  for (int c = 0; c < 32; ++c) a2[c] = 0.0f;
  for (int head = 0; head < 8; ++head) {
    float lg[4], oxv[4], oyv[4];
    for (int k = 0; k < 4; ++k) {
      int c2 = (head * 4 + k) * 2;
      float ox = boff[c2], oy = boff[c2 + 1], lgt = batt[head * 4 + k];
      for (int c = 0; c < 32; ++c) {
        ox += ql[c] * Woff[c * 64 + c2];
        oy += ql[c] * Woff[c * 64 + c2 + 1];
        lgt += ql[c] * Watt[c * 32 + head * 4 + k];
      }
      oxv[k] = ox; oyv[k] = oy; lg[k] = lgt;
    }
    float mx = fmaxf(fmaxf(lg[0], lg[1]), fmaxf(lg[2], lg[3]));
    float es[4], sum = 0.0f;
    for (int k = 0; k < 4; ++k) { es[k] = expf(lg[k] - mx); sum += es[k]; }
    float inv = 1.0f / sum;
    const float* vch = vimg + (size_t)(bn * 32 + head * 4) * HW_;
    for (int k = 0; k < 4; ++k) {
      float aw = es[k] * inv;
      float ppx = (float)w + oxv[k] * (79.0f / 80.0f);
      float ppy = (float)h + oyv[k] * (79.0f / 80.0f);
      float y0f = floorf(ppy), x0f = floorf(ppx);
      int iy0 = (int)y0f, ix0 = (int)x0f;
      float wy = ppy - y0f, wx = ppx - x0f;
      float w00 = (1 - wy) * (1 - wx), w01 = (1 - wy) * wx;
      float w10 = wy * (1 - wx), w11 = wy * wx;
      bool vy0 = (iy0 >= 0 && iy0 < H_), vy1 = (iy0 + 1 >= 0 && iy0 + 1 < H_);
      bool vx0 = (ix0 >= 0 && ix0 < W_), vx1 = (ix0 + 1 >= 0 && ix0 + 1 < W_);
      int cy0 = max(0, min(H_ - 1, iy0)), cy1 = max(0, min(H_ - 1, iy0 + 1));
      int cx0 = max(0, min(W_ - 1, ix0)), cx1 = max(0, min(W_ - 1, ix0 + 1));
      for (int d = 0; d < 4; ++d) {
        const float* ch = vch + (size_t)d * HW_;
        float s = 0.0f;
        if (vy0 && vx0) s += ch[cy0 * W_ + cx0] * w00;
        if (vy0 && vx1) s += ch[cy0 * W_ + cx1] * w01;
        if (vy1 && vx0) s += ch[cy1 * W_ + cx0] * w10;
        if (vy1 && vx1) s += ch[cy1 * W_ + cx1] * w11;
        a2[head * 4 + d] += aw * s;
      }
    }
  }
  float* orow = a2out + (size_t)gid * 32;
  for (int c = 0; c < 32; ++c) {
    float acc = bout2[c];
    for (int d = 0; d < 32; ++d) acc += a2[d] * Wout2[d * 32 + c];
    orow[c] = acc;
  }
}

// ---------------- 6) x3 (frame-concat bug) -> q3/k3/v3 projections ----------
__global__ void k_x3proj(const float* __restrict__ a2out,
                         const float* __restrict__ Wq3, const float* __restrict__ bq3,
                         const float* __restrict__ Wk3, const float* __restrict__ bk3,
                         const float* __restrict__ Wv3, const float* __restrict__ bv3,
                         float* __restrict__ q3cm, _Float16* __restrict__ q3h,
                         float* __restrict__ kfcm, float* __restrict__ vfcm) {
  int gid = blockIdx.x * blockDim.x + threadIdx.x;
  if (gid >= B_ * NTOK_) return;
  int b = gid / NTOK_;
  int t = gid - b * NTOK_;
  int n = t / HW_;
  int hw = t - n * HW_;
  const int srcmap[4] = {0, 0, 1, 2};  // faithful frame-concat bug
  int src = srcmap[n];
  const float* xr = a2out + (((size_t)b * NIMG_ + src) * HW_ + hw) * 32;
  float xv[32];
  for (int c = 0; c < 32; ++c) xv[c] = xr[c];
  for (int o = 0; o < 32; ++o) {
    float qv = bq3[o], kv = bk3[o], vv = bv3[o];
    const float* wq = Wq3 + o * 32;
    const float* wk = Wk3 + o * 32;
    const float* wv = Wv3 + o * 32;
    for (int c = 0; c < 32; ++c) {
      qv += wq[c] * xv[c];
      kv += wk[c] * xv[c];
      vv += wv[c] * xv[c];
    }
    size_t cm = (size_t)b * 32 * NTOK_ + (size_t)o * NTOK_ + t;
    q3cm[cm] = qv;
    kfcm[cm] = kv;
    vfcm[cm] = vv;
    q3h[((size_t)b * NTOK_ + t) * 32 + o] = (_Float16)qv;
  }
}

// ---------------- 7a) strided conv3d for offsets + gelu ---------------------
__global__ void k_hid(const float* __restrict__ q3cm, const float* __restrict__ WA,
                      const float* __restrict__ bA, float* __restrict__ hid) {
  int gid = blockIdx.x * blockDim.x + threadIdx.x;
  if (gid >= B_ * 32 * LR_) return;
  int b = gid / (32 * LR_);
  int rem = gid - b * 32 * LR_;
  int oc = rem / LR_;
  int l = rem - oc * LR_;
  int zo = l / 100, r2 = l - zo * 100;
  int yo = r2 / 10, xo = r2 - yo * 10;
  float acc = bA[oc];
  const float* qb = q3cm + (size_t)b * 32 * NTOK_;
  for (int ic = 0; ic < 32; ++ic) {
    const float* chp = qb + (size_t)ic * NTOK_;
    const float* wp = WA + ((size_t)oc * 32 + ic) * 400;
    for (int kd = 0; kd < 4; ++kd) {
      int z = zo * 2 - 1 + kd;
      if (z < 0 || z > 3) continue;
      for (int kh = 0; kh < 10; ++kh) {
        int y = yo * 8 - 1 + kh;
        if (y < 0 || y >= H_) continue;
        const float* rowp = chp + z * HW_ + y * W_;
        const float* wr = wp + kd * 100 + kh * 10;
        for (int kw = 0; kw < 10; ++kw) {
          int xx = xo * 8 - 1 + kw;
          if (xx < 0 || xx >= W_) continue;
          acc += rowp[xx] * wr[kw];
        }
      }
    }
  }
  hid[((size_t)b * 32 + oc) * LR_ + l] = geluf(acc);
}

// ---------------- 7b) offsets -> tanh*scale -> sample coords ----------------
__global__ void k_off3(const float* __restrict__ hid, const float* __restrict__ WB,
                       const float* __restrict__ bB, float* __restrict__ czb,
                       float* __restrict__ cyb, float* __restrict__ cxb) {
  int gid = blockIdx.x * blockDim.x + threadIdx.x;
  if (gid >= B_ * LR_) return;
  int b = gid / LR_;
  int l = gid - b * LR_;
  float o3c[3];
  const float osc[3] = {2.0f, 8.0f, 8.0f};
  for (int comp = 0; comp < 3; ++comp) {
    float s = bB[comp];
    for (int oc = 0; oc < 32; ++oc) s += WB[comp * 32 + oc] * hid[((size_t)b * 32 + oc) * LR_ + l];
    o3c[comp] = tanhf(s) * osc[comp];
  }
  int zo = l / 100, r2 = l - zo * 100;
  int yo = r2 / 10, xo = r2 - yo * 10;
  czb[gid] = (float)zo * 2.0f + 0.5f + o3c[0];
  cyb[gid] = (float)yo * 8.0f + 3.5f + o3c[1];
  cxb[gid] = (float)xo * 8.0f + 3.5f + o3c[2];
}

// ---------------- 7c) trilinear sample K/V at 200 points -> f16 padded ------
__global__ void k_sample3d(const float* __restrict__ kfcm, const float* __restrict__ vfcm,
                           const float* __restrict__ czb, const float* __restrict__ cyb,
                           const float* __restrict__ cxb, _Float16* __restrict__ ksh,
                           _Float16* __restrict__ vsh) {
  int gid = blockIdx.x * blockDim.x + threadIdx.x;
  if (gid >= B_ * LP_ * 32) return;
  int b = gid / (LP_ * 32);
  int rem = gid - b * LP_ * 32;
  int l = rem / 32;
  int c = rem - l * 32;
  size_t oidx = ((size_t)b * LP_ + l) * 32 + c;
  if (l >= LR_) { ksh[oidx] = (_Float16)0.0f; vsh[oidx] = (_Float16)0.0f; return; }
  float z = czb[b * LR_ + l], y = cyb[b * LR_ + l], x = cxb[b * LR_ + l];
  float z0f = floorf(z), y0f = floorf(y), x0f = floorf(x);
  int z0 = (int)z0f, y0 = (int)y0f, x0 = (int)x0f;
  float wz = z - z0f, wy = y - y0f, wx = x - x0f;
  const float* kc_ = kfcm + (size_t)b * 32 * NTOK_ + (size_t)c * NTOK_;
  const float* vc_ = vfcm + (size_t)b * 32 * NTOK_ + (size_t)c * NTOK_;
  float ka = 0.0f, va = 0.0f;
  for (int dz = 0; dz < 2; ++dz)
    for (int dy = 0; dy < 2; ++dy)
      for (int dx = 0; dx < 2; ++dx) {
        int zi = z0 + dz, yi = y0 + dy, xi = x0 + dx;
        if (zi < 0 || zi > NIMG_ - 1 || yi < 0 || yi >= H_ || xi < 0 || xi >= W_) continue;
        float wgt = (dz ? wz : 1 - wz) * (dy ? wy : 1 - wy) * (dx ? wx : 1 - wx);
        int off = zi * HW_ + yi * W_ + xi;
        ka += kc_[off] * wgt;
        va += vc_[off] * wgt;
      }
  ksh[oidx] = (_Float16)ka;
  vsh[oidx] = (_Float16)va;
}

// ---------------- 8) fused WMMA attention: softmax(QK^T/sqrt(32)) V ---------
__global__ __launch_bounds__(32) void k_attn3d(const _Float16* __restrict__ q3h,
                                               const _Float16* __restrict__ ksh,
                                               const _Float16* __restrict__ vsh,
                                               _Float16* __restrict__ o3h) {
  __shared__ __align__(16) _Float16 ksl[LP_ * 32];
  __shared__ __align__(16) _Float16 vsl[LP_ * 32];
  __shared__ float scb[16 * LP_];
  __shared__ _Float16 phb[16 * LP_];
  const int lane = threadIdx.x;
  const int bid = blockIdx.x;
  const int b = bid / (NTOK_ / 16);
  const int tile = bid - b * (NTOK_ / 16);
  const int row0 = tile * 16;
  const _Float16* ksg = ksh + (size_t)b * LP_ * 32;
  const _Float16* vsg = vsh + (size_t)b * LP_ * 32;
#if defined(STDA_ASYNC_LDS)
  // CDNA5 async global->LDS b128 copies, tracked by ASYNCcnt.
  for (int i = lane; i < (LP_ * 32) / 8; i += 32) {
    __builtin_amdgcn_global_load_async_to_lds_b128(
        (as1_v4i*)(uintptr_t)(ksg + i * 8),
        (as3_v4i*)(uint32_t)(uintptr_t)(ksl + i * 8), 0, 0);
    __builtin_amdgcn_global_load_async_to_lds_b128(
        (as1_v4i*)(uintptr_t)(vsg + i * 8),
        (as3_v4i*)(uint32_t)(uintptr_t)(vsl + i * 8), 0, 0);
  }
  __builtin_amdgcn_s_wait_asynccnt(0);
#else
  {
    const uint4* ks4 = (const uint4*)ksg;
    const uint4* vs4 = (const uint4*)vsg;
    uint4* ksl4 = (uint4*)ksl;
    uint4* vsl4 = (uint4*)vsl;
    for (int i = lane; i < (LP_ * 32) / 8; i += 32) {
      ksl4[i] = ks4[i];
      vsl4[i] = vs4[i];
    }
  }
#endif
  __syncthreads();
  const int m16 = lane & 15;
  v16h a;
  const _Float16* qrow = q3h + ((size_t)b * NTOK_ + row0 + m16) * 32;
  for (int j = 0; j < 16; ++j) a[j] = qrow[amapA(lane, j)];
  const int mb = (lane >> 4) * 8;
  for (int nt = 0; nt < LP_ / 16; ++nt) {
    v16h bf;
    for (int j = 0; j < 16; ++j) bf[j] = ksl[(nt * 16 + m16) * 32 + bmapK(lane, j)];
    v8f accz;
    for (int e = 0; e < 8; ++e) accz[e] = 0.0f;
    accz = wmma16(a, bf, accz);
    for (int rI = 0; rI < 8; ++rI) scb[(mb + rI) * LP_ + nt * 16 + m16] = accz[rI];
  }
  __syncthreads();
  if (lane < 16) {
    const float sc_ = 0.17677669529663687f;  // 1/sqrt(32)
    float mx = -1e30f;
    for (int l = 0; l < LR_; ++l) {
      float v = scb[lane * LP_ + l] * sc_;
      if (v > mx) mx = v;
    }
    float sum = 0.0f;
    for (int l = 0; l < LR_; ++l) {
      float e = expf(scb[lane * LP_ + l] * sc_ - mx);
      scb[lane * LP_ + l] = e;
      sum += e;
    }
    float inv = 1.0f / sum;
    for (int l = 0; l < LP_; ++l)
      phb[lane * LP_ + l] = (l < LR_) ? (_Float16)(scb[lane * LP_ + l] * inv) : (_Float16)0.0f;
  }
  __syncthreads();
  v8f acc0, acc1;
  for (int e = 0; e < 8; ++e) { acc0[e] = 0.0f; acc1[e] = 0.0f; }
  for (int kc = 0; kc < LP_ / 32; ++kc) {
    v16h pa, bv0, bv1;
    for (int j = 0; j < 16; ++j) pa[j] = phb[m16 * LP_ + kc * 32 + amapA(lane, j)];
    for (int j = 0; j < 16; ++j) {
      int kk = kc * 32 + bmapK(lane, j);
      bv0[j] = vsl[kk * 32 + m16];
      bv1[j] = vsl[kk * 32 + 16 + m16];
    }
    acc0 = wmma16(pa, bv0, acc0);
    acc1 = wmma16(pa, bv1, acc1);
  }
  for (int rI = 0; rI < 8; ++rI) {
    _Float16* orow = o3h + ((size_t)b * NTOK_ + row0 + mb + rI) * 32;
    orow[m16] = (_Float16)acc0[rI];
    orow[16 + m16] = (_Float16)acc1[rI];
  }
}

// ---------------- 9) final WMMA GEMM: out = rate1*(o3 W_out3^T + b) + conv --
__global__ __launch_bounds__(32) void k_outproj(const _Float16* __restrict__ o3h,
                                                const _Float16* __restrict__ wout3h,
                                                const float* __restrict__ bout3,
                                                const float* __restrict__ rate1p,
                                                float* __restrict__ out) {
  const int lane = threadIdx.x;
  const int row0 = blockIdx.x * 16;
  const int m16 = lane & 15;
  v16h a;
  const _Float16* arow = o3h + (size_t)(row0 + m16) * 32;
  for (int j = 0; j < 16; ++j) a[j] = arow[amapA(lane, j)];
  const float r1 = rate1p[0];
  const int bidx = row0 / NTOK_;
  const int tbase = row0 - bidx * NTOK_;
  const int mb = (lane >> 4) * 8;
  for (int nt = 0; nt < 16; ++nt) {
    v16h bw;
    const int col = nt * 16 + m16;
    for (int j = 0; j < 16; ++j) bw[j] = wout3h[col * 32 + bmapK(lane, j)];
    v8f acc;
    for (int e = 0; e < 8; ++e) acc[e] = 0.0f;
    acc = wmma16(a, bw, acc);
    const float bo = bout3[col];
    for (int rI = 0; rI < 8; ++rI) {
      int t = tbase + mb + rI;
      size_t oi = ((size_t)t * B_ + bidx) * C_ + col;
      out[oi] = r1 * (acc[rI] + bo) + out[oi];  // out already holds rate2*conv
    }
  }
}

// ---------------------------------------------------------------------------
extern "C" void kernel_launch(void* const* d_in, const int* in_sizes, int n_in,
                              void* d_out, int out_size, void* d_ws, size_t ws_size,
                              hipStream_t stream) {
  (void)in_sizes; (void)n_in; (void)out_size; (void)ws_size;
  const float* x      = (const float*)d_in[0];
  const float* W_qkv  = (const float*)d_in[1];
  const float* b_qkv  = (const float*)d_in[2];
  const float* W_fc   = (const float*)d_in[3];
  const float* b_fc   = (const float*)d_in[4];
  const float* W_doff = (const float*)d_in[5];
  const float* b_doff = (const float*)d_in[6];
  const float* W_dep  = (const float*)d_in[7];
  const float* b_dep  = (const float*)d_in[8];
  const float* W_st   = (const float*)d_in[9];
  const float* b_st   = (const float*)d_in[10];
  const float* W_off2d= (const float*)d_in[11];
  const float* b_off2d= (const float*)d_in[12];
  const float* W_attw = (const float*)d_in[13];
  const float* b_attw = (const float*)d_in[14];
  const float* W_val  = (const float*)d_in[15];
  const float* b_val  = (const float*)d_in[16];
  const float* W_out2d= (const float*)d_in[17];
  const float* b_out2d= (const float*)d_in[18];
  const float* W_q3   = (const float*)d_in[19];
  const float* b_q3   = (const float*)d_in[20];
  const float* W_k3   = (const float*)d_in[21];
  const float* b_k3   = (const float*)d_in[22];
  const float* W_v3   = (const float*)d_in[23];
  const float* b_v3   = (const float*)d_in[24];
  const float* W_offA = (const float*)d_in[25];
  const float* b_offA = (const float*)d_in[26];
  const float* W_offB = (const float*)d_in[27];
  const float* b_offB = (const float*)d_in[28];
  const float* W_out3 = (const float*)d_in[29];
  const float* b_out3 = (const float*)d_in[30];
  const float* rate1  = (const float*)d_in[31];
  const float* rate2  = (const float*)d_in[32];
  float* out = (float*)d_out;

  char* base = (char*)d_ws;
  size_t off = 0;
  auto alloc = [&](size_t bytes) -> char* {
    char* p = base + off;
    off += (bytes + 255) & ~(size_t)255;
    return p;
  };
  _Float16* wqkvh  = (_Float16*)alloc((size_t)256 * 64 * 2);
  _Float16* wout3h = (_Float16*)alloc((size_t)256 * 32 * 2);
  float* qkv   = (float*)alloc((size_t)B_ * NTOK_ * 64 * 4);     // 26.2 MB
  float* fin   = (float*)alloc((size_t)16 * 36 * HW_ * 4);       // 14.7 MB
  float* offb  = (float*)alloc((size_t)16 * 18 * HW_ * 4);       //  7.4 MB
  float* a2out = (float*)alloc((size_t)B_ * NTOK_ * 32 * 4);     // 13.1 MB
  float* q3cm  = (float*)alloc((size_t)B_ * 32 * NTOK_ * 4);     // 13.1 MB
  float* kfcm  = (float*)alloc((size_t)B_ * 32 * NTOK_ * 4);     // 13.1 MB
  float* vfcm  = (float*)alloc((size_t)B_ * 32 * NTOK_ * 4);     // 13.1 MB
  _Float16* q3h = (_Float16*)alloc((size_t)B_ * NTOK_ * 32 * 2); //  6.6 MB
  _Float16* o3h = (_Float16*)alloc((size_t)B_ * NTOK_ * 32 * 2); //  6.6 MB
  float* hid = (float*)alloc((size_t)B_ * 32 * LR_ * 4);
  float* czb = (float*)alloc((size_t)B_ * LR_ * 4);
  float* cyb = (float*)alloc((size_t)B_ * LR_ * 4);
  float* cxb = (float*)alloc((size_t)B_ * LR_ * 4);
  _Float16* ksh = (_Float16*)alloc((size_t)B_ * LP_ * 32 * 2);
  _Float16* vsh = (_Float16*)alloc((size_t)B_ * LP_ * 32 * 2);
  float* vimg = fin;  // alias: fin is dead after k_depst (stream-ordered)

  k_wconv<<<64, 256, 0, stream>>>(W_qkv, W_out3, wqkvh, wout3h);
  k_qkv<<<(B_ * NTOK_) / 16, 32, 0, stream>>>(x, wqkvh, b_qkv, qkv);
  k_fin<<<(B_ * NIMG_ * HW_ + 255) / 256, 256, 0, stream>>>(qkv, W_fc, b_fc, fin);
  k_offconv<<<(16 * 18 * HW_ + 255) / 256, 256, 0, stream>>>(fin, W_doff, b_doff, offb);
  k_depst<<<(B_ * NIMG_ * HW_ * 4 + 255) / 256, 256, 0, stream>>>(
      fin, offb, W_dep, b_dep, W_st, b_st, rate2, out);
  k_vimg<<<(B_ * NIMG_ * HW_ + 255) / 256, 256, 0, stream>>>(qkv, W_val, b_val, vimg);
  k_attn2d<<<(B_ * NIMG_ * HW_ + 255) / 256, 256, 0, stream>>>(
      qkv, vimg, W_off2d, b_off2d, W_attw, b_attw, W_out2d, b_out2d, a2out);
  k_x3proj<<<(B_ * NTOK_ + 255) / 256, 256, 0, stream>>>(
      a2out, W_q3, b_q3, W_k3, b_k3, W_v3, b_v3, q3cm, q3h, kfcm, vfcm);
  k_hid<<<(B_ * 32 * LR_ + 255) / 256, 256, 0, stream>>>(q3cm, W_offA, b_offA, hid);
  k_off3<<<(B_ * LR_ + 255) / 256, 256, 0, stream>>>(hid, W_offB, b_offB, czb, cyb, cxb);
  k_sample3d<<<(B_ * LP_ * 32 + 255) / 256, 256, 0, stream>>>(
      kfcm, vfcm, czb, cyb, cxb, ksh, vsh);
  k_attn3d<<<B_ * (NTOK_ / 16), 32, 0, stream>>>(q3h, ksh, vsh, o3h);
  k_outproj<<<(B_ * NTOK_) / 16, 32, 0, stream>>>(o3h, wout3h, b_out3, rate1, out);
}
